// ComplexEnzymeModel_10187662426919
// MI455X (gfx1250) — matrix operand order repaired and checked by reading, not compiled, not run on hardware
//
#include <hip/hip_runtime.h>

typedef _Float16 h16;
typedef __attribute__((ext_vector_type(16))) _Float16 v16h;
typedef __attribute__((ext_vector_type(8)))  _Float16 v8h;
typedef __attribute__((ext_vector_type(8)))  float    v8f;

#define NN 100000
#define EE 3200000
#define GG 512
#define HH 64
#define CC 7

// ---------------- init: deg=1 (self loop), pooled=0, cnt=0 ----------------
__global__ __launch_bounds__(256) void k_init(float* deg, float* pooled, float* cnt) {
    int t = blockIdx.x * 256 + threadIdx.x;
    if (t < NN) deg[t] = 1.0f;
    if (t < GG * HH) pooled[t] = 0.0f;
    if (t < GG) cnt[t] = 0.0f;
}

// ---------------- degree over edges ----------------
__global__ __launch_bounds__(256) void k_deg(const int* __restrict__ ei, float* __restrict__ deg) {
    int e = blockIdx.x * 256 + threadIdx.x;
    if (e >= EE) return;
    atomicAdd(&deg[ei[EE + e]], 1.0f);   // dst side
}

// ---------------- dis = rsqrt(deg) in place; xa = dis^2 * x (self loop term) ----------------
__global__ __launch_bounds__(256) void k_dis_xa(float* __restrict__ dis, const float* __restrict__ x,
                                                float* __restrict__ xa) {
    int n = blockIdx.x * 256 + threadIdx.x;
    if (n >= NN) return;
    float dv = rsqrtf(dis[n]);   // deg >= 1 always (self loop)
    dis[n] = dv;
    xa[n] = dv * dv * x[n];
}

// ---------------- scalar edge aggregation for layer 1 ----------------
__global__ __launch_bounds__(256) void k_xa_edges(const int* __restrict__ ei, const float* __restrict__ dis,
                                                  const float* __restrict__ x, float* __restrict__ xa) {
    int e = blockIdx.x * 256 + threadIdx.x;
    if (e >= EE) return;
    int s = ei[e], d = ei[EE + e];
    atomicAdd(&xa[d], dis[s] * dis[d] * x[s]);
}

// ---------------- h1 = relu(xa * W1 + b1), stored f16 row-major [N,64] ----------------
__global__ __launch_bounds__(256) void k_h1(const float* __restrict__ xa, const float* __restrict__ W1,
                                            const float* __restrict__ b1, h16* __restrict__ h1h) {
    int t = blockIdx.x * 256 + threadIdx.x;
    if (t >= NN * HH) return;
    int n = t >> 6, j = t & 63;
    float v = fmaxf(xa[n] * W1[j] + b1[j], 0.0f);
    h1h[t] = (h16)v;
}

// ---------------- pack W2 into WMMA B-operand lane layout ----------------
// layout: [kchunk(2)][coltile(4)][lane(32)][16 halves]; lane<16 -> K=kb+0..7,kb+16..23 col=lane,
// lane>=16 -> K=kb+8..15,kb+24..31 (kb = 32*kchunk), col = lane-16 (+16*coltile)
__global__ __launch_bounds__(256) void k_packW2(const float* __restrict__ W2, h16* __restrict__ w2p) {
    int t = threadIdx.x;                 // one block of 256
    int lane = t & 31;
    int ctc = t >> 5;                    // c*4 + ct
    int c = ctc >> 2, ct = ctc & 3;
    int n = ct * 16 + (lane & 15);
    int kb = c * 32 + ((lane >> 4) << 3);
    for (int i = 0; i < 16; ++i)
        w2p[t * 16 + i] = (h16)W2[(kb + i + ((i >> 3) << 3)) * HH + n];
}

// ---------------- h2pre = h1 @ W2 via v_wmma_f32_16x16x32_f16 ----------------
__global__ __launch_bounds__(256) void k_gemm_wmma(const h16* __restrict__ h1h, const h16* __restrict__ w2p,
                                                   float* __restrict__ h2pre) {
    int wave = (blockIdx.x * 256 + threadIdx.x) >> 5;    // one 16-row tile per wave
    int lane = threadIdx.x & 31;
    const int ntile = NN / 16;                            // 6250, exact
    if (wave >= ntile) return;                            // uniform per wave -> EXEC all ones

    const h16* arow = h1h + (size_t)(wave * 16 + (lane & 15)) * HH + ((lane >> 4) << 3);
    v8h lo0 = *(const v8h*)(arow + 0);
    v8h hi0 = *(const v8h*)(arow + 16);
    v8h lo1 = *(const v8h*)(arow + 32);
    v8h hi1 = *(const v8h*)(arow + 48);
    v16h a0 = __builtin_shufflevector(lo0, hi0, 0,1,2,3,4,5,6,7,8,9,10,11,12,13,14,15);
    v16h a1 = __builtin_shufflevector(lo1, hi1, 0,1,2,3,4,5,6,7,8,9,10,11,12,13,14,15);

    int rbase = wave * 16 + ((lane >> 4) << 3);
#pragma unroll
    for (int ct = 0; ct < 4; ++ct) {
        v16h b0 = *(const v16h*)(w2p + ((size_t)(0 * 4 + ct) * 32 + lane) * 16);
        v16h b1 = *(const v16h*)(w2p + ((size_t)(1 * 4 + ct) * 32 + lane) * 16);
        v8f acc = {};
        acc = __builtin_amdgcn_wmma_f32_16x16x32_f16(false, a0, false, b0, (short)0, acc, false, false);
        acc = __builtin_amdgcn_wmma_f32_16x16x32_f16(false, a1, false, b1, (short)0, acc, false, false);
        int col = ct * 16 + (lane & 15);
#pragma unroll
        for (int r = 0; r < 8; ++r)
            h2pre[(size_t)(rbase + r) * HH + col] = acc[r];
    }
}

// ---------------- agg2 init with self-loop term: agg2 = dis^2 * h2pre ----------------
__global__ __launch_bounds__(256) void k_agg2_init(const float* __restrict__ dis, const float* __restrict__ h2pre,
                                                   float* __restrict__ agg2) {
    int t = blockIdx.x * 256 + threadIdx.x;
    if (t >= NN * HH) return;
    int n = t >> 6;
    agg2[t] = dis[n] * dis[n] * h2pre[t];
}

// ---------------- layer-2 edge aggregation: 16 threads/edge, float4 gather + 4 atomics ----------------
__global__ __launch_bounds__(256) void k_agg2_edges(const int* __restrict__ ei, const float* __restrict__ dis,
                                                    const float* __restrict__ h2pre, float* __restrict__ agg2) {
    unsigned t = blockIdx.x * 256u + threadIdx.x;
    if (t >= (unsigned)EE * 16u) return;
    int e = (int)(t >> 4);
    int q = ((int)t & 15) << 2;
    int s = ei[e], d = ei[EE + e];
    float nrm = dis[s] * dis[d];
    const float4 v = *(const float4*)(h2pre + (size_t)s * HH + q);
    float* o = agg2 + (size_t)d * HH + q;
    atomicAdd(o + 0, nrm * v.x);
    atomicAdd(o + 1, nrm * v.y);
    atomicAdd(o + 2, nrm * v.z);
    atomicAdd(o + 3, nrm * v.w);
}

// ---------------- h2 = relu(agg2 + b2) fused with mean-pool accumulation ----------------
__global__ __launch_bounds__(256) void k_h2_pool(const float* __restrict__ agg2, const float* __restrict__ b2,
                                                 const int* __restrict__ batch, float* __restrict__ pooled,
                                                 float* __restrict__ cnt) {
    int t = blockIdx.x * 256 + threadIdx.x;
    if (t >= NN * HH) return;
    int n = t >> 6, j = t & 63;
    float v = fmaxf(agg2[t] + b2[j], 0.0f);
    int g = batch[n];
    atomicAdd(&pooled[g * HH + j], v);
    if (j == 0) atomicAdd(&cnt[g], 1.0f);
}

// ---------------- head: one wave per graph, 64->32 relu, 32->7 via LDS ----------------
__global__ __launch_bounds__(256) void k_head(const float* __restrict__ pooled, const float* __restrict__ cnt,
                                              const float* __restrict__ fW1, const float* __restrict__ fb1,
                                              const float* __restrict__ fW2, const float* __restrict__ fb2,
                                              float* __restrict__ out) {
    __shared__ float hid[8][32];
    int lane = threadIdx.x & 31;
    int w = threadIdx.x >> 5;
    int g = blockIdx.x * 8 + w;          // 64 blocks * 8 = 512 graphs, exact
    float inv = 1.0f / fmaxf(cnt[g], 1.0f);
    float acc = fb1[lane];
#pragma unroll 8
    for (int k = 0; k < HH; ++k)
        acc += pooled[g * HH + k] * inv * fW1[k * 32 + lane];
    hid[w][lane] = fmaxf(acc, 0.0f);
    __syncthreads();
    if (lane < CC) {
        float o = fb2[lane];
#pragma unroll
        for (int k = 0; k < 32; ++k) o += hid[w][k] * fW2[k * CC + lane];
        out[g * CC + lane] = o;
    }
}

extern "C" void kernel_launch(void* const* d_in, const int* in_sizes, int n_in,
                              void* d_out, int out_size, void* d_ws, size_t ws_size,
                              hipStream_t stream) {
    const float* x   = (const float*)d_in[0];
    const int*   ei  = (const int*)d_in[1];     // [2,E]: src = ei[0:E), dst = ei[E:2E)
    const int*   bat = (const int*)d_in[2];
    const float* W1  = (const float*)d_in[3];
    const float* b1  = (const float*)d_in[4];
    const float* W2  = (const float*)d_in[5];
    const float* b2  = (const float*)d_in[6];
    const float* fW1 = (const float*)d_in[7];
    const float* fb1 = (const float*)d_in[8];
    const float* fW2 = (const float*)d_in[9];
    const float* fb2 = (const float*)d_in[10];
    float* out = (float*)d_out;

    // workspace carve-up (all offsets multiples of 256 B)
    const int NPAD = (NN + 63) & ~63;           // 100032
    float* deg    = (float*)d_ws;               // N floats; becomes dis in place
    float* xa     = deg + NPAD;
    h16*   h1h    = (h16*)(xa + NPAD);          // N*64 halves (12.8 MB)
    h16*   w2p    = h1h + (size_t)NN * HH;      // 4096 halves
    float* h2pre  = (float*)(w2p + 4096);       // N*64 floats (25.6 MB)
    float* agg2   = h2pre + (size_t)NN * HH;    // N*64 floats (25.6 MB)
    float* pooled = agg2 + (size_t)NN * HH;     // G*64
    float* cnt    = pooled + GG * HH;           // G
    float* dis    = deg;

    const int TB = 256;
    int gN    = (NN + TB - 1) / TB;             // 391
    int gE    = (EE + TB - 1) / TB;             // 12500
    int gNH   = (NN * HH + TB - 1) / TB;        // 25000
    unsigned gE16 = ((unsigned)EE * 16u + TB - 1) / TB;  // 200000
    int gGemm = (NN / 16 + 7) / 8;              // 8 waves/block -> 782

    k_init<<<gN, TB, 0, stream>>>(deg, pooled, cnt);
    k_deg<<<gE, TB, 0, stream>>>(ei, deg);
    k_dis_xa<<<gN, TB, 0, stream>>>(dis, x, xa);
    k_xa_edges<<<gE, TB, 0, stream>>>(ei, dis, x, xa);
    k_h1<<<gNH, TB, 0, stream>>>(xa, W1, b1, h1h);
    k_packW2<<<1, TB, 0, stream>>>(W2, w2p);
    k_gemm_wmma<<<gGemm, TB, 0, stream>>>(h1h, w2p, h2pre);
    k_agg2_init<<<gNH, TB, 0, stream>>>(dis, h2pre, agg2);
    k_agg2_edges<<<gE16, TB, 0, stream>>>(ei, dis, h2pre, agg2);
    k_h2_pool<<<gNH, TB, 0, stream>>>(agg2, b2, bat, pooled, cnt);
    k_head<<<GG / 8, TB, 0, stream>>>(pooled, cnt, fW1, fb1, fW2, fb2, out);
}
